// EALSTM_15195594293501
// MI455X (gfx1250) — matrix operand order repaired and checked by reading, not compiled, not run on hardware
//
#include <hip/hip_runtime.h>
#include <hip/hip_bf16.h>

typedef __attribute__((ext_vector_type(16))) _Float16 v16h;
typedef __attribute__((ext_vector_type(8)))  float    v8f;
typedef __attribute__((ext_vector_type(4)))  float    v4f;
typedef __attribute__((ext_vector_type(2)))  float    v2f;

#define DDYN   3
#define HDIM   256
#define TSTEPS 365
#define BATCH  1024
#define N3     768   // 3*H (f|g|o)
#define BT     32    // batch rows per workgroup
#define MT     2     // BT/16 batch tiles
#define NTL    3     // 16-wide n-tiles per wave (48 columns)
#define KT     8     // k-tiles of 32 (K=256)
#define KREG   6     // k-tiles with register-resident A; kt=6,7 live in LDS

struct Smem {
  float    z[BT * N3];                   // z^T as [b][n], f32           (96 KB)
  _Float16 hfrag[KT * MT * 32 * 16];     // h^T WMMA B fragments         (16 KB)
  _Float16 wlds[16 * 2 * NTL * 32 * 16]; // kt=6,7 A fragments per wave  (96 KB)
  float    ig[BT * HDIM];                // static input gate i          (32 KB)
  float    cst[BT * HDIM];               // cell state c                 (32 KB)
  float    xs[BT * 4];                   // x_dyn tile for current t
  float    red[512];                     // head reduction
};

__device__ __forceinline__ float fast_sigmoid(float x) {
  x = fminf(fmaxf(x, -30.f), 30.f);
  return __builtin_amdgcn_rcpf(1.0f + __expf(-x));   // v_rcp_f32
}
__device__ __forceinline__ float fast_tanh(float x) {
  x = fminf(fmaxf(x, -15.f), 15.f);
  float e = __expf(2.0f * x);
  return (e - 1.0f) * __builtin_amdgcn_rcpf(e + 1.0f);
}

__global__ __launch_bounds__(512, 4)   // <=256 VGPR: full 16-wave residency
void ealstm_kernel(const float* __restrict__ x_dyn, const float* __restrict__ x_stat,
                   const float* __restrict__ w_i,  const float* __restrict__ b_i,
                   const float* __restrict__ w_f,  const float* __restrict__ b_f,
                   const float* __restrict__ w_g,  const float* __restrict__ b_g,
                   const float* __restrict__ w_o,  const float* __restrict__ b_o,
                   const float* __restrict__ w_head, const float* __restrict__ b_head,
                   float* __restrict__ out)
{
  extern __shared__ char smem_raw[];
  Smem& sm = *reinterpret_cast<Smem*>(smem_raw);

  const int tid  = threadIdx.x;
  const int wv   = tid >> 5;      // wave 0..15 -> z columns [48*wv, 48*wv+48)
  const int lane = tid & 31;
  const int lh   = lane >> 4;
  const int ln   = lane & 15;
  const int b0   = blockIdx.x * BT;

  // ---------------- one-time staging ----------------
  for (int idx = tid; idx < KT*MT*32*16; idx += 512)
    sm.hfrag[idx] = (_Float16)0.f;                    // h0 = 0

  const int ubl = tid >> 4;          // update phase: batch row 0..31
  const int jb  = (tid & 15) << 4;   // hidden base 0,16..240
  {
    const int bg = b0 + ubl;
    const float s0 = x_stat[bg*3+0], s1 = x_stat[bg*3+1], s2 = x_stat[bg*3+2];
    #pragma unroll
    for (int q = 0; q < 16; ++q) {
      const int j = jb + q;
      float a = b_i[j] + s0*w_i[0*HDIM+j] + s1*w_i[1*HDIM+j] + s2*w_i[2*HDIM+j];
      sm.ig[ubl*HDIM + j] = fast_sigmoid(a);
      sm.cst[ubl*HDIM + j] = 0.f;                     // c0 = 0
    }
  }

  // ---------------- Wh^T -> f16 WMMA A fragments (kt 0..5 regs, kt 6,7 LDS) -----
  // 16-bit A 16x32, this code's k order (B same): lanes 0-15 (row M=ln) hold K=0..15; lanes 16-31 K=16..31
  v16h wf[NTL][KREG];
  v2f  wxA[NTL];   // f32 16x16x4 A: [Wx0 Wx1 Wx2 bias] columns
  const int wbase = (wv*2*NTL*32 + lane)*16;          // this wave's LDS A region
  #pragma unroll
  for (int ntl = 0; ntl < NTL; ++ntl) {
    const int n = wv*48 + ntl*16 + ln;
    const int gate = n >> 8, col = n & 255;
    const float* W  = (gate == 0) ? w_f : (gate == 1) ? w_g : w_o;
    const float* Bv = (gate == 0) ? b_f : (gate == 1) ? b_g : b_o;
    #pragma unroll
    for (int kt = 0; kt < KT; ++kt) {
      v16h f;
      #pragma unroll
      for (int hi = 0; hi < 16; ++hi) {
        const int K = hi + (lh << 4);
        f[hi] = (_Float16)W[(DDYN + kt*32 + K)*HDIM + col];
      }
      if (kt < KREG) wf[ntl][kt] = f;
      else *(v16h*)&sm.wlds[wbase + ((kt-KREG)*NTL + ntl)*32*16] = f; // -> LDS
    }
    // f32 A 16x4 layout: VGPR0 = K0|K2, VGPR1 = K1|K3 (lanes 0-15 | 16-31)
    wxA[ntl][0] = lh ? W[2*HDIM + col] : W[0*HDIM + col];
    wxA[ntl][1] = lh ? Bv[col]         : W[1*HDIM + col];
  }

  __syncthreads();

  // this thread's h write slot (B-fragment layout: lane = 16*(kk/16)+b%16, half = kk%16)
  const int u_kt   = jb >> 5;
  const int u_lane = (((jb & 31) >> 4) << 4) + (ubl & 15);
  const int u_mt   = ubl >> 4;
  _Float16* const hw = &sm.hfrag[((u_kt*MT + u_mt)*32 + u_lane)*16];

  // ---------------- recurrence ----------------
  for (int t = 0; t < TSTEPS; ++t) {
    if (tid < BT*3) {                       // stage x_dyn[:, t, :]
      const int b = tid / 3, d = tid - b*3;
      sm.xs[b*4 + d] = x_dyn[(size_t)(b0 + b)*(TSTEPS*DDYN) + t*DDYN + d];
    }
    __syncthreads();                        // xs ready, hfrag(t-1) ready

    // z^T = Wh^T @ h^T + [Wx|b]^T @ [x|1]  — all through the matrix pipe
    #pragma unroll
    for (int mt = 0; mt < MT; ++mt) {
      const int bl = mt*16 + ln;
      // f32 B 4x16: VGPR0 = K0|K2, VGPR1 = K1|K3 -> [x0 x1 x2 1] per batch col
      v2f xB;
      xB[0] = lh ? sm.xs[bl*4+2] : sm.xs[bl*4+0];
      xB[1] = lh ? 1.0f          : sm.xs[bl*4+1];

      v8f zacc = {};
      v8f acc[NTL];
      #pragma unroll
      for (int ntl = 0; ntl < NTL; ++ntl)
        acc[ntl] = __builtin_amdgcn_wmma_f32_16x16x4_f32(
            false, wxA[ntl], false, xB, (short)0, zacc, false, false);

      const int hb = (mt*32 + lane)*16;     // B fragment base; kt stride 1024 halves
      v16h bfr = *(const v16h*)&sm.hfrag[hb];
      v16h aw6[NTL], aw7[NTL];              // LDS-resident kt=6,7 A fragments
      #pragma unroll
      for (int kt = 0; kt < KT; ++kt) {
        v16h bnx;
        if (kt + 1 < KT)                    // double buffer B
          bnx = *(const v16h*)&sm.hfrag[hb + (kt+1)*MT*32*16];
        if (kt == 4) {                      // prefetch kt=6 A tile 2 groups ahead
          #pragma unroll
          for (int ntl = 0; ntl < NTL; ++ntl)
            aw6[ntl] = *(const v16h*)&sm.wlds[wbase + (0*NTL + ntl)*32*16];
        }
        if (kt == 5) {                      // prefetch kt=7 A tile 2 groups ahead
          #pragma unroll
          for (int ntl = 0; ntl < NTL; ++ntl)
            aw7[ntl] = *(const v16h*)&sm.wlds[wbase + (1*NTL + ntl)*32*16];
        }
        #pragma unroll
        for (int ntl = 0; ntl < NTL; ++ntl) {
          const v16h a = (kt < KREG) ? wf[ntl][kt] : (kt == KREG ? aw6[ntl] : aw7[ntl]);
          acc[ntl] = __builtin_amdgcn_wmma_f32_16x16x32_f16(
              false, a, false, bfr, (short)0, acc[ntl], false, false);
        }
        if (kt + 1 < KT) bfr = bnx;
      }

      #pragma unroll
      for (int ntl = 0; ntl < NTL; ++ntl) {
        const int nb = wv*48 + ntl*16 + (lh << 3);
        float* zp = &sm.z[bl*N3 + nb];
        v4f lo, hi2;
        lo[0]=acc[ntl][0]; lo[1]=acc[ntl][1]; lo[2]=acc[ntl][2]; lo[3]=acc[ntl][3];
        hi2[0]=acc[ntl][4]; hi2[1]=acc[ntl][5]; hi2[2]=acc[ntl][6]; hi2[3]=acc[ntl][7];
        *(v4f*)(zp)     = lo;
        *(v4f*)(zp + 4) = hi2;
      }
    }
    __syncthreads();                        // z ready; hfrag free to overwrite

    // gates: c = sig(zf)*c + i*tanh(zg); h = sig(zo)*tanh(c)  (2 chunks of 8)
    {
      const float* zrow = &sm.z[ubl*N3 + jb];
      float*       crow = &sm.cst[ubl*HDIM + jb];
      const float* irow = &sm.ig[ubl*HDIM + jb];
      v16h hv;
      #pragma unroll
      for (int half = 0; half < 2; ++half) {
        const int off = half*8;
        v4f F[2], G[2], O[2], C4[2], I4[2];
        #pragma unroll
        for (int r = 0; r < 2; ++r) {
          F[r]  = *(const v4f*)(zrow + off + 4*r);
          G[r]  = *(const v4f*)(zrow + HDIM   + off + 4*r);
          O[r]  = *(const v4f*)(zrow + 2*HDIM + off + 4*r);
          C4[r] = *(const v4f*)(crow + off + 4*r);
          I4[r] = *(const v4f*)(irow + off + 4*r);
        }
        #pragma unroll
        for (int r = 0; r < 2; ++r) {
          #pragma unroll
          for (int e = 0; e < 4; ++e) {
            const float f = fast_sigmoid(F[r][e]);
            const float g = fast_tanh(G[r][e]);
            const float o = fast_sigmoid(O[r][e]);
            const float c = f*C4[r][e] + I4[r][e]*g;
            C4[r][e] = c;
            hv[off + r*4 + e] = (_Float16)(o * fast_tanh(c));
          }
        }
        #pragma unroll
        for (int r = 0; r < 2; ++r) *(v4f*)(crow + off + 4*r) = C4[r];
      }
      *(v16h*)hw = hv;                      // 32B contiguous
    }
  }

  // ---------------- head: out = h @ w_head + b_head ----------------
  {
    const v16h hv = *(const v16h*)hw;
    float p = 0.f;
    #pragma unroll
    for (int q = 0; q < 16; ++q) p += (float)hv[q] * w_head[jb + q];
    sm.red[tid] = p;
  }
  __syncthreads();
  if (tid < BT) {
    float s = 0.f;
    #pragma unroll
    for (int k = 0; k < 16; ++k) s += sm.red[tid*16 + k];
    out[b0 + tid] = s + b_head[0];
  }
}

extern "C" void kernel_launch(void* const* d_in, const int* in_sizes, int n_in,
                              void* d_out, int out_size, void* d_ws, size_t ws_size,
                              hipStream_t stream) {
  const float* x_dyn  = (const float*)d_in[0];
  const float* x_stat = (const float*)d_in[1];
  const float* w_i    = (const float*)d_in[2];
  const float* b_i    = (const float*)d_in[3];
  const float* w_f    = (const float*)d_in[4];
  const float* b_f    = (const float*)d_in[5];
  const float* w_g    = (const float*)d_in[6];
  const float* b_g    = (const float*)d_in[7];
  const float* w_o    = (const float*)d_in[8];
  const float* b_o    = (const float*)d_in[9];
  const float* w_head = (const float*)d_in[10];
  const float* b_head = (const float*)d_in[11];
  float* out = (float*)d_out;

  dim3 grid(BATCH / BT), block(512);
  ealstm_kernel<<<grid, block, sizeof(Smem), stream>>>(
      x_dyn, x_stat, w_i, b_i, w_f, b_f, w_g, b_g, w_o, b_o,
      w_head, b_head, out);
}
